// SelfAttentionLayer_54254026883596
// MI455X (gfx1250) — matrix operand order repaired and checked by reading, not compiled, not run on hardware
//
#include <hip/hip_runtime.h>

typedef __attribute__((ext_vector_type(16))) _Float16 v16h;
typedef __attribute__((ext_vector_type(8)))  _Float16 v8h;
typedef __attribute__((ext_vector_type(8)))  float    v8f;

#define SEQ 8192
#define CTX 2048
#define DK  128
#define ATTN_SCALE 11.313708498984761f   // sqrt(128): reference MULTIPLIES by sqrt(dk)

// ---------------------------------------------------------------------------
// WMMA fragment loaders following the CDNA5 ISA VGPR layouts (CDNA5 ISA):
//  A (16x32 f16, M x K): lane r (r=lane&15) holds row M=r.
//    lane half h=0: K = {0..7, 16..23}; h=1: K = {8..15, 24..31}
//  B (32x16 f16, K x N): lane n (lane&15) holds column N=n.
//    h=0: K = {0..7, 16..23}; h=1: K = {8..15, 24..31}   (two B128 loads)
//  C/D (16x16 f32): VGPR j, lane<16 -> M=j, lane>=16 -> M=j+8 ; N = lane&15
// ---------------------------------------------------------------------------
static __device__ __forceinline__ v16h load_a_f16(const _Float16* __restrict__ src,
                                                  int ld, int row0, int k0) {
  const int lane = threadIdx.x & 31;
  const int r = lane & 15, h = lane >> 4;
  const _Float16* p = src + (size_t)(row0 + r) * ld + k0 + h * 8;
  v8h lo = *(const v8h*)(p);        // K = k0 + h*8 + [0..7]
  v8h hi = *(const v8h*)(p + 16);   // K = k0 + 16 + h*8 + [0..7]
  return __builtin_shufflevector(lo, hi, 0,1,2,3,4,5,6,7,8,9,10,11,12,13,14,15);
}

// B column n comes from row (n0+n) of a row-major matrix: B[kk][n] = mat[n0+n][k0+kk]
static __device__ __forceinline__ v16h load_b_f16(const _Float16* __restrict__ mat,
                                                  int ld, int n0, int k0) {
  const int lane = threadIdx.x & 31;
  const int n = lane & 15, h = lane >> 4;
  const _Float16* p = mat + (size_t)(n0 + n) * ld + k0 + h * 8;
  v8h lo = *(const v8h*)(p);
  v8h hi = *(const v8h*)(p + 16);
  return __builtin_shufflevector(lo, hi, 0,1,2,3,4,5,6,7,8,9,10,11,12,13,14,15);
}

static __device__ __forceinline__ v8f wmma_f16(v16h a, v16h b, v8f c) {
  return __builtin_amdgcn_wmma_f32_16x16x32_f16(false, a, false, b, (short)0, c,
                                                false, false);
}

// gfx1250 async copy global -> LDS, 16B per lane, tracked by ASYNCcnt.
static __device__ __forceinline__ void async_cp16(unsigned lds_off, const void* g) {
  unsigned long long ga = (unsigned long long)(size_t)g;
  asm volatile("global_load_async_to_lds_b128 %0, %1, off"
               :: "v"(lds_off), "v"(ga) : "memory");
}

static __device__ __forceinline__ void wg_barrier() {
  asm volatile("s_barrier_signal -1\n\ts_barrier_wait -1" ::: "memory");
}

// ---------------------------------------------------------------------------
// Kernel 1: f32 -> f16 convert (x), 4 elements/thread
// ---------------------------------------------------------------------------
__global__ void k_cvt_f16(const float* __restrict__ in, _Float16* __restrict__ out, int n) {
  int i = (blockIdx.x * blockDim.x + threadIdx.x) * 4;
  if (i + 3 < n) {
    float4 f = *(const float4*)(in + i);
    out[i + 0] = (_Float16)f.x;
    out[i + 1] = (_Float16)f.y;
    out[i + 2] = (_Float16)f.z;
    out[i + 3] = (_Float16)f.w;
  }
}

// Kernel 2: W [CTX][DK] f32 -> Wt [DK][CTX] f16 (so GEMM B-fragments are contiguous)
__global__ void k_cvt_wT(const float* __restrict__ w, _Float16* __restrict__ wT) {
  int i = blockIdx.x * blockDim.x + threadIdx.x;   // over CTX*DK
  int c = i >> 7, d = i & 127;
  wT[(size_t)d * CTX + c] = (_Float16)w[i];
}

// ---------------------------------------------------------------------------
// Kernel 3: fused QKV projection, software-pipelined over the K(=CTX) loop so
// next-iteration fragment loads overlap the current WMMA chain.
// block = 256 thr (8 waves) computes a 64(s) x 128(d) tile; grid.y = proj id.
// proj 0/1 -> Q/K row-major f16; proj 2 -> Vt = V^T [DK][SEQ] f16.
// ---------------------------------------------------------------------------
__global__ void k_project(const _Float16* __restrict__ xh,
                          const _Float16* __restrict__ wT3,   // [3][DK][CTX]
                          _Float16* __restrict__ qk,          // Q then K, each [SEQ][DK]
                          _Float16* __restrict__ vt) {        // [DK][SEQ]
  const int proj = blockIdx.y;
  const _Float16* w = wT3 + (size_t)proj * DK * CTX;
  const int wave = threadIdx.x >> 5;
  const int row0 = blockIdx.x * 64 + (wave & 3) * 16;
  const int col0 = (wave >> 2) * 64;

  v8f acc[4] = {};
  v16h a = load_a_f16(xh, CTX, row0, 0);
  v16h b[4];
#pragma unroll
  for (int i = 0; i < 4; ++i) b[i] = load_b_f16(w, CTX, col0 + 16 * i, 0);

  for (int k0 = 0; k0 < CTX - 32; k0 += 32) {
    v16h an = load_a_f16(xh, CTX, row0, k0 + 32);   // prefetch next fragments
    v16h bn[4];
#pragma unroll
    for (int i = 0; i < 4; ++i) bn[i] = load_b_f16(w, CTX, col0 + 16 * i, k0 + 32);
#pragma unroll
    for (int i = 0; i < 4; ++i) acc[i] = wmma_f16(a, b[i], acc[i]);
    a = an;
#pragma unroll
    for (int i = 0; i < 4; ++i) b[i] = bn[i];
  }
#pragma unroll
  for (int i = 0; i < 4; ++i) acc[i] = wmma_f16(a, b[i], acc[i]);  // epilogue step

  const int lane = threadIdx.x & 31;
  const int n = lane & 15, h = lane >> 4;
  if (proj < 2) {
    _Float16* dst = qk + (size_t)proj * SEQ * DK;
#pragma unroll
    for (int i = 0; i < 4; ++i)
#pragma unroll
      for (int j = 0; j < 8; ++j) {
        int s = row0 + j + h * 8;
        int d = col0 + 16 * i + n;
        dst[(size_t)s * DK + d] = (_Float16)acc[i][j];
      }
  } else {
#pragma unroll
    for (int i = 0; i < 4; ++i)
#pragma unroll
      for (int j = 0; j < 8; ++j) {
        int s = row0 + j + h * 8;
        int d = col0 + 16 * i + n;
        vt[(size_t)d * SEQ + s] = (_Float16)acc[i][j];
      }
  }
}

// ---------------------------------------------------------------------------
// Kernel 4: column softmax stats.  For each t: m_t = max_s score, Z_t = sum exp.
// Wave owns 16 t-columns (B frags held in regs).  The Q s-tile (16x128 = 4KB)
// is identical for all 8 waves -> staged into LDS with async copies,
// double-buffered with s_wait_asynccnt + split barriers.
// ---------------------------------------------------------------------------
__global__ void __launch_bounds__(256) k_colstats(
    const _Float16* __restrict__ Q, const _Float16* __restrict__ K,
    float* __restrict__ mOut, float* __restrict__ rZOut) {
  __shared__ __align__(16) _Float16 ldsQ[2][16 * DK];   // 2 x 4KB

  const int tid = threadIdx.x;
  const int wave = tid >> 5;
  const int t0 = blockIdx.x * 128 + wave * 16;

  v16h bk[4];
#pragma unroll
  for (int c = 0; c < 4; ++c) bk[c] = load_b_f16(K, DK, t0, 32 * c);

  // prologue: stage s-tile 0 into buffer 0 (1 x b128 per thread = 4KB)
  async_cp16((unsigned)(size_t)&ldsQ[0][0] + tid * 16,
             (const char*)Q + tid * 16);

  float m = -3.0e38f, Z = 0.0f;
  for (int it = 0; it < SEQ / 16; ++it) {
    const int buf = it & 1;
    const int sn = (it + 1 < SEQ / 16) ? (it + 1) * 16 : 0;  // dummy on last iter
    async_cp16((unsigned)(size_t)&ldsQ[buf ^ 1][0] + tid * 16,
               (const char*)(Q + (size_t)sn * DK) + tid * 16);
    asm volatile("s_wait_asynccnt 0x1" ::: "memory");  // prior buffer landed (mine)
    wg_barrier();                                      // ... and everyone else's

    v8f acc = {};
#pragma unroll
    for (int c = 0; c < 4; ++c)
      acc = wmma_f16(load_a_f16(&ldsQ[buf][0], DK, 0, 32 * c), bk[c], acc);

    float tm = acc[0] * ATTN_SCALE;
#pragma unroll
    for (int j = 1; j < 8; ++j) tm = fmaxf(tm, acc[j] * ATTN_SCALE);
    float mnew = fmaxf(m, tm);
    float zadd = 0.0f;
#pragma unroll
    for (int j = 0; j < 8; ++j) zadd += __expf(acc[j] * ATTN_SCALE - mnew);
    Z = Z * __expf(m - mnew) + zadd;
    m = mnew;

    wg_barrier();   // all waves done reading buf before next writes land in it
  }

  float om = __shfl_xor(m, 16, 32);
  float oZ = __shfl_xor(Z, 16, 32);
  float mc = fmaxf(m, om);
  float Zc = Z * __expf(m - mc) + oZ * __expf(om - mc);

  const int lane = threadIdx.x & 31;
  if (lane < 16) {
    int t = t0 + lane;
    mOut[t] = mc;
    rZOut[t] = 1.0f / Zc;
  }
}

// ---------------------------------------------------------------------------
// Kernel 5: out[s,:] = sum_t exp(scale*q_s.k_t - m_t)/Z_t * v[t,:]
// K tile (32x128) and V tile (128x32) are identical for all 8 waves ->
// async-staged into LDS, double-buffered.  Per 32-wide t step: 8 score WMMAs,
// exp/normalize, P-tile transpose bounce through per-wave LDS (s_wait_dscnt),
// 8 PV WMMAs.
// ---------------------------------------------------------------------------
__global__ void __launch_bounds__(256) k_attnout(
    const _Float16* __restrict__ Q, const _Float16* __restrict__ K,
    const _Float16* __restrict__ Vt, const float* __restrict__ mIn,
    const float* __restrict__ rZIn, float* __restrict__ out) {
  __shared__ __align__(16) _Float16 ldsK[2][32 * DK];   // 2 x 8KB, row-major ld=DK
  __shared__ __align__(16) _Float16 ldsV[2][DK * 32];   // 2 x 8KB, [d][tt] ld=32
  __shared__ __align__(16) _Float16 ldsP[8][16 * 32];   // per-wave P tile, 8KB

  const int tid = threadIdx.x;
  const int wave = tid >> 5;
  const int lane = tid & 31;
  const int n = lane & 15, h = lane >> 4;
  const int s0 = blockIdx.x * 128 + wave * 16;

  v16h aq[4];
#pragma unroll
  for (int c = 0; c < 4; ++c) aq[c] = load_a_f16(Q, DK, s0, 32 * c);

  v8f acc[8] = {};
  _Float16* lp = &ldsP[wave][0];

  // ---- cooperative async stage of K/V tiles for t-block `t` into buffer `bf`
  auto issue_tiles = [&](int bf, int t) {
    const char* gk = (const char*)(K + (size_t)t * DK);     // contiguous 8KB
    unsigned lk = (unsigned)(size_t)&ldsK[bf][0];
#pragma unroll
    for (int r = 0; r < 2; ++r)
      async_cp16(lk + tid * 16 + r * 4096, gk + tid * 16 + r * 4096);
    unsigned lv = (unsigned)(size_t)&ldsV[bf][0];           // 128 rows x 64B
#pragma unroll
    for (int r = 0; r < 2; ++r) {
      int c = tid + r * 256;
      int row = c >> 2, cc = c & 3;
      async_cp16(lv + row * 64 + cc * 16,
                 (const char*)(Vt + (size_t)row * SEQ + t) + cc * 16);
    }
  };

  issue_tiles(0, 0);
  for (int it = 0; it < SEQ / 32; ++it) {
    const int t0 = it * 32;
    const int buf = it & 1;
    const int tn = (t0 + 32 < SEQ) ? t0 + 32 : 0;   // dummy issue on last iter
    issue_tiles(buf ^ 1, tn);
    asm volatile("s_wait_asynccnt 0x4" ::: "memory");  // my prior 4 copies done
    wg_barrier();                                      // everyone's copies done

    const _Float16* kt = &ldsK[buf][0];
    const _Float16* vv = &ldsV[buf][0];

    v8f sc0 = {}, sc1 = {};
#pragma unroll
    for (int c = 0; c < 4; ++c) {
      sc0 = wmma_f16(aq[c], load_b_f16(kt, DK, 0, 32 * c), sc0);
      sc1 = wmma_f16(aq[c], load_b_f16(kt, DK, 16, 32 * c), sc1);
    }

    float m0 = mIn[t0 + n],      r0 = rZIn[t0 + n];
    float m1 = mIn[t0 + 16 + n], r1 = rZIn[t0 + 16 + n];

#pragma unroll
    for (int j = 0; j < 8; ++j) {
      int row = j + h * 8;
      lp[row * 32 + n]      = (_Float16)(__expf(sc0[j] * ATTN_SCALE - m0) * r0);
      lp[row * 32 + 16 + n] = (_Float16)(__expf(sc1[j] * ATTN_SCALE - m1) * r1);
    }
    asm volatile("s_wait_dscnt 0x0" ::: "memory");   // wave-local P RAW

    v16h ap = load_a_f16(lp, 32, 0, 0);   // 16(s) x 32(t) probability tile
#pragma unroll
    for (int d = 0; d < 8; ++d)
      acc[d] = wmma_f16(ap, load_b_f16(vv, 32, 16 * d, 0), acc[d]);

    wg_barrier();   // all waves done reading buf before next writes land in it
  }

#pragma unroll
  for (int d = 0; d < 8; ++d)
#pragma unroll
    for (int j = 0; j < 8; ++j) {
      int s = s0 + j + h * 8;
      out[(size_t)s * DK + 16 * d + n] = acc[d][j];
    }
}

// ---------------------------------------------------------------------------
extern "C" void kernel_launch(void* const* d_in, const int* in_sizes, int n_in,
                              void* d_out, int out_size, void* d_ws, size_t ws_size,
                              hipStream_t stream) {
  (void)in_sizes; (void)n_in; (void)out_size; (void)ws_size;

  const float* x  = (const float*)d_in[0];
  const float* wq = (const float*)d_in[1];
  const float* wk = (const float*)d_in[2];
  const float* wv = (const float*)d_in[3];
  float* out = (float*)d_out;

  char* ws = (char*)d_ws;
  size_t off = 0;
  auto carve = [&](size_t bytes) -> void* {
    void* p = ws + off;
    off += (bytes + 255) & ~(size_t)255;
    return p;
  };
  _Float16* xh  = (_Float16*)carve((size_t)SEQ * CTX * sizeof(_Float16));
  _Float16* wT  = (_Float16*)carve((size_t)3 * DK * CTX * sizeof(_Float16));
  _Float16* Qh  = (_Float16*)carve((size_t)2 * SEQ * DK * sizeof(_Float16)); // Q then K
  _Float16* VtH = (_Float16*)carve((size_t)DK * SEQ * sizeof(_Float16));
  float* mBuf   = (float*)carve((size_t)SEQ * sizeof(float));
  float* rZBuf  = (float*)carve((size_t)SEQ * sizeof(float));
  _Float16* Kh  = Qh + (size_t)SEQ * DK;

  // 1) convert x to f16; transpose-convert the three W's
  k_cvt_f16<<<(SEQ * CTX) / 1024, 256, 0, stream>>>(x, xh, SEQ * CTX);
  k_cvt_wT<<<(CTX * DK) / 256, 256, 0, stream>>>(wq, wT + 0 * (size_t)DK * CTX);
  k_cvt_wT<<<(CTX * DK) / 256, 256, 0, stream>>>(wk, wT + 1 * (size_t)DK * CTX);
  k_cvt_wT<<<(CTX * DK) / 256, 256, 0, stream>>>(wv, wT + 2 * (size_t)DK * CTX);

  // 2) fused QKV projection (WMMA GEMM, pipelined); V written transposed
  k_project<<<dim3(SEQ / 64, 3), 256, 0, stream>>>(xh, wT, Qh, VtH);

  // 3) column-wise softmax stats over s (online max/sum-exp, async-staged Q)
  k_colstats<<<SEQ / 128, 256, 0, stream>>>(Qh, Kh, mBuf, rZBuf);

  // 4) second pass: recompute scores, normalize, P @ V (async-staged K/V)
  k_attnout<<<SEQ / 128, 256, 0, stream>>>(Qh, Kh, VtH, mBuf, rZBuf, out);
}
